// CrossAttentionBlock_21277267985025
// MI455X (gfx1250) — compile-verified, matrix operand order checked by reading, not run on hardware
//
#include <hip/hip_runtime.h>
#include <hip/hip_bf16.h>

#define N_SPATIAL 4096   // 64*64
#define CDIM 256
#define NBATCH 4

typedef __attribute__((ext_vector_type(16))) __bf16 bf16x16;
typedef __attribute__((ext_vector_type(8)))  __bf16 bf16x8;
typedef __attribute__((ext_vector_type(8)))  float  f32x8;

// float -> bf16 with round-to-nearest-even (bit manipulation; no cvt dependence)
__device__ __forceinline__ __bf16 f2bf(float f) {
    unsigned u = __float_as_uint(f);
    u += 0x7FFFu + ((u >> 16) & 1u);
    unsigned short s = (unsigned short)(u >> 16);
    return __builtin_bit_cast(__bf16, s);
}

__device__ __forceinline__ f32x8 wmma_bf16(bf16x16 a, bf16x16 b, f32x8 c) {
    return __builtin_amdgcn_wmma_f32_16x16x32_bf16(
        /*neg_a=*/false, a, /*neg_b=*/false, b,
        /*c_mod=*/(short)0, c, /*reuse_a=*/false, /*reuse_b=*/false);
}

// Fragment loader: elems 0..7 from p[0..7], elems 8..15 from p[16..23].
// Caller passes p = base + 8*half; this matches the 16-bit A/B VGPR layout
// (lanes 0-15: K=0..7 & 16..23 ; lanes 16-31: K=8..15 & 24..31).
__device__ __forceinline__ bf16x16 load_frag(const __bf16* p) {
    bf16x8 a = *(const bf16x8*)p;
    bf16x8 b = *(const bf16x8*)(p + 16);
    bf16x16 f;
#pragma unroll
    for (int i = 0; i < 8; ++i) { f[i] = a[i]; f[i + 8] = b[i]; }
    return f;
}

__device__ __forceinline__ f32x8 zero8() {
    f32x8 c;
#pragma unroll
    for (int i = 0; i < 8; ++i) c[i] = 0.0f;
    return c;
}

// Async copy of 16B from global to LDS (tracked by ASYNCcnt).
// lds_addr = wave-relative LDS byte offset (low 32 bits of generic pointer).
__device__ __forceinline__ void async_b128(unsigned lds_addr, const void* gptr) {
    unsigned long long ga = (unsigned long long)(size_t)gptr;
    asm volatile("global_load_async_to_lds_b128 %0, %1, off"
                 :: "v"(lds_addr), "v"(ga) : "memory");
}
__device__ __forceinline__ void wait_async0() {
    asm volatile("s_wait_asynccnt 0x0" ::: "memory");
}

// ---------------- LayerNorm: one block (256 threads) per spatial position ----
__global__ void ln_kernel(const float* __restrict__ x, const float* __restrict__ gamma,
                          const float* __restrict__ beta, float* __restrict__ xn,
                          __bf16* __restrict__ xnb) {
    __shared__ float red[256];
    size_t row = blockIdx.x;
    int t = threadIdx.x;
    float v = x[row * CDIM + t];
    red[t] = v; __syncthreads();
#pragma unroll
    for (int s = 128; s > 0; s >>= 1) { if (t < s) red[t] += red[t + s]; __syncthreads(); }
    float mu = red[0] * (1.0f / CDIM);
    __syncthreads();
    float d = v - mu;
    red[t] = d * d; __syncthreads();
#pragma unroll
    for (int s = 128; s > 0; s >>= 1) { if (t < s) red[t] += red[t + s]; __syncthreads(); }
    float var = red[0] * (1.0f / CDIM);
    float y = d * rsqrtf(var + 1e-3f) * gamma[t] + beta[t];
    xn[row * CDIM + t]  = y;
    xnb[row * CDIM + t] = f2bf(y);
}

// ---------------- fp32 -> bf16 elementwise ----------------------------------
__global__ void cvt_bf16_kernel(const float* __restrict__ src, __bf16* __restrict__ dst, int n) {
    int i = blockIdx.x * blockDim.x + threadIdx.x;
    if (i < n) dst[i] = f2bf(src[i]);
}

// ---------------- transpose 4 weight matrices to [N][K] bf16 ----------------
__global__ void transpose_w_kernel(const float* __restrict__ Wq, const float* __restrict__ Wk,
                                   const float* __restrict__ Wv, const float* __restrict__ Wp,
                                   __bf16* __restrict__ WqT, __bf16* __restrict__ WkT,
                                   __bf16* __restrict__ WvT, __bf16* __restrict__ WpT) {
    int i = blockIdx.x * blockDim.x + threadIdx.x;  // 0..262143
    int w = i >> 16;
    int j = i & 65535;
    int k = j >> 8, n = j & 255;
    const float* S = (w == 0) ? Wq : (w == 1) ? Wk : (w == 2) ? Wv : Wp;
    __bf16* D      = (w == 0) ? WqT : (w == 1) ? WkT : (w == 2) ? WvT : WpT;
    D[n * 256 + k] = f2bf(S[k * 256 + n]);
}

// ---------------- Q projection: wave = 16x16 output tile --------------------
__global__ void qproj_kernel(const __bf16* __restrict__ A, const __bf16* __restrict__ WT,
                             const float* __restrict__ bias, __bf16* __restrict__ Qo) {
    int wid  = (blockIdx.x * blockDim.x + threadIdx.x) >> 5;
    int lane = threadIdx.x & 31;
    int half = lane >> 4, r = lane & 15;
    size_t m0 = (size_t)(wid >> 4) * 16;
    int n0 = (wid & 15) * 16;
    const __bf16* arow = A  + (m0 + r) * CDIM + 8 * half;
    const __bf16* brow = WT + (size_t)(n0 + r) * CDIM + 8 * half;
    f32x8 c = zero8();
#pragma unroll
    for (int kk = 0; kk < 8; ++kk)
        c = wmma_bf16(load_frag(arow + kk * 32), load_frag(brow + kk * 32), c);
    float bv = bias[n0 + r];
#pragma unroll
    for (int i = 0; i < 8; ++i)
        Qo[(m0 + i + 8 * half) * CDIM + n0 + r] = f2bf(c[i] + bv);
}

// ---------------- K,V projection; V stored transposed [B][C][N] -------------
__global__ void kvproj_kernel(const __bf16* __restrict__ A, const __bf16* __restrict__ WkT,
                              const __bf16* __restrict__ WvT, const float* __restrict__ bk,
                              const float* __restrict__ bvv, __bf16* __restrict__ Ko,
                              __bf16* __restrict__ VTo) {
    int wid  = (blockIdx.x * blockDim.x + threadIdx.x) >> 5;
    int lane = threadIdx.x & 31;
    int half = lane >> 4, r = lane & 15;
    size_t m0 = (size_t)(wid >> 4) * 16;
    int n0 = (wid & 15) * 16;
    const __bf16* arow  = A   + (m0 + r) * CDIM + 8 * half;
    const __bf16* bkrow = WkT + (size_t)(n0 + r) * CDIM + 8 * half;
    const __bf16* bvrow = WvT + (size_t)(n0 + r) * CDIM + 8 * half;
    f32x8 ck = zero8(), cv = zero8();
#pragma unroll
    for (int kk = 0; kk < 8; ++kk) {
        bf16x16 af = load_frag(arow + kk * 32);
        ck = wmma_bf16(af, load_frag(bkrow + kk * 32), ck);
        cv = wmma_bf16(af, load_frag(bvrow + kk * 32), cv);
    }
    float kb = bk[n0 + r];
#pragma unroll
    for (int i = 0; i < 8; ++i)
        Ko[(m0 + i + 8 * half) * CDIM + n0 + r] = f2bf(ck[i] + kb);
    // V transposed store: lane's 8 C-layout rows are consecutive key indices.
    int b   = (int)(m0 >> 12);                 // 4096 keys per batch
    int nk0 = (int)(m0 & 4095) + 8 * half;
    float vb = bvv[n0 + r];
    bf16x8 pack;
#pragma unroll
    for (int i = 0; i < 8; ++i) pack[i] = f2bf(cv[i] + vb);
    *(bf16x8*)(VTo + ((size_t)b * CDIM + n0 + r) * N_SPATIAL + nk0) = pack;
}

// ---------------- Flash attention ------------------------------------------
// Block = 8 waves, each wave owns a 16-row query tile; the 32-key K/V chunk is
// staged once per block into LDS via async global->LDS copies and shared.
__global__ void __launch_bounds__(256) flash_kernel(
        const __bf16* __restrict__ Q, const __bf16* __restrict__ K,
        const __bf16* __restrict__ VT, __bf16* __restrict__ O) {
    __shared__ __bf16 Kc[32][256];     // 16 KB: chunk rows of K (row-major)
    __shared__ __bf16 Vc[256][32];     // 16 KB: V^T columns for this chunk
    __shared__ __bf16 Pst[8][16 * 32]; // 8 KB: per-wave P staging

    int t    = threadIdx.x;
    int wib  = t >> 5;
    int wid  = blockIdx.x * 8 + wib;
    int lane = t & 31;
    int half = lane >> 4, r = lane & 15;
    int b  = wid >> 8;                 // 256 query tiles per batch
    int m0 = (wid & 255) * 16;

    const __bf16* Qb = Q  + ((size_t)b * N_SPATIAL) * CDIM;
    const __bf16* Kb = K  + ((size_t)b * N_SPATIAL) * CDIM;
    const __bf16* Vb = VT + (size_t)b * CDIM * N_SPATIAL;

    bf16x16 qf[8];
    const __bf16* qrow = Qb + (size_t)(m0 + r) * CDIM + 8 * half;
#pragma unroll
    for (int kk = 0; kk < 8; ++kk) qf[kk] = load_frag(qrow + kk * 32);

    f32x8 o[16];
#pragma unroll
    for (int u = 0; u < 16; ++u) o[u] = zero8();
    float m[8], l[8];
#pragma unroll
    for (int i = 0; i < 8; ++i) { m[i] = -1e30f; l[i] = 0.0f; }

    __bf16* pst = &Pst[wib][0];
    const unsigned kl = (unsigned)(size_t)&Kc[0][0];  // LDS byte offsets
    const unsigned vl = (unsigned)(size_t)&Vc[0][0];
    const float sc = 0.0625f;                         // 256^-0.5

    for (int j0 = 0; j0 < N_SPATIAL; j0 += 32) {
        __syncthreads();   // previous chunk's compute finished in all waves
        // ---- stage K chunk (contiguous 16 KB) and V chunk into LDS ----
        const char* kgb = (const char*)Kb + (size_t)j0 * (CDIM * 2);
#pragma unroll
        for (int i = 0; i < 4; ++i) {
            int li = i * 256 + t;                     // 0..1023 -> 16B segments
            async_b128(kl + li * 16, kgb + li * 16);
        }
#pragma unroll
        for (int i = 0; i < 4; ++i) {
            int li = i * 256 + t;
            int u = li >> 2, seg = li & 3;
            async_b128(vl + li * 16, Vb + (size_t)u * N_SPATIAL + j0 + seg * 8);
        }
        wait_async0();
        __syncthreads();   // chunk visible to all waves

        // ---- S = Q @ K_chunk^T : two 16x16 tiles, fragments from LDS ----
        f32x8 s0 = zero8(), s1 = zero8();
        const __bf16* k0base = &Kc[r][8 * half];
        const __bf16* k1base = &Kc[16 + r][8 * half];
#pragma unroll
        for (int kk = 0; kk < 8; ++kk) {
            s0 = wmma_bf16(qf[kk], load_frag(k0base + kk * 32), s0);
            s1 = wmma_bf16(qf[kk], load_frag(k1base + kk * 32), s1);
        }
        // ---- online softmax (row stats local to the 16-lane half) ----
        float p0[8], p1[8], mn[8];
#pragma unroll
        for (int i = 0; i < 8; ++i) {
            p0[i] = s0[i] * sc; p1[i] = s1[i] * sc;
            mn[i] = fmaxf(p0[i], p1[i]);
        }
#pragma unroll
        for (int off = 8; off >= 1; off >>= 1)
#pragma unroll
            for (int i = 0; i < 8; ++i)
                mn[i] = fmaxf(mn[i], __shfl_xor(mn[i], off, 32));
        float alpha[8], ls[8];
#pragma unroll
        for (int i = 0; i < 8; ++i) {
            float mnew = fmaxf(m[i], mn[i]);
            alpha[i] = __expf(m[i] - mnew);
            m[i] = mnew;
            p0[i] = __expf(p0[i] - mnew);
            p1[i] = __expf(p1[i] - mnew);
            ls[i] = p0[i] + p1[i];
        }
#pragma unroll
        for (int off = 8; off >= 1; off >>= 1)
#pragma unroll
            for (int i = 0; i < 8; ++i)
                ls[i] += __shfl_xor(ls[i], off, 32);
#pragma unroll
        for (int i = 0; i < 8; ++i) l[i] = l[i] * alpha[i] + ls[i];
#pragma unroll
        for (int u = 0; u < 16; ++u)
#pragma unroll
            for (int i = 0; i < 8; ++i) o[u][i] *= alpha[i];

        // ---- stage P (C layout) -> A-layout fragment via per-wave LDS ----
#pragma unroll
        for (int i = 0; i < 8; ++i) {
            int row = i + 8 * half;
            pst[row * 32 + r]      = f2bf(p0[i]);
            pst[row * 32 + 16 + r] = f2bf(p1[i]);
        }
        bf16x8 pa = *(const bf16x8*)&pst[r * 32 + 8 * half];
        bf16x8 pb = *(const bf16x8*)&pst[r * 32 + 8 * half + 16];
        bf16x16 pf;
#pragma unroll
        for (int i = 0; i < 8; ++i) { pf[i] = pa[i]; pf[i + 8] = pb[i]; }

        // ---- O += P @ V_chunk (B fragments from LDS, contiguous) ----
        const __bf16* vbase = &Vc[r][8 * half];
#pragma unroll
        for (int u = 0; u < 16; ++u)
            o[u] = wmma_bf16(pf, load_frag(vbase + u * 16 * 32), o[u]);
    }

    // epilogue: divide by softmax denominator, store bf16 row-major
    float inv[8];
#pragma unroll
    for (int i = 0; i < 8; ++i) inv[i] = 1.0f / l[i];
    __bf16* Ob = O + ((size_t)b * N_SPATIAL + m0) * CDIM;
#pragma unroll
    for (int u = 0; u < 16; ++u)
#pragma unroll
        for (int i = 0; i < 8; ++i)
            Ob[(size_t)(i + 8 * half) * CDIM + u * 16 + r] = f2bf(o[u][i] * inv[i]);
}

// ---------------- Output projection + bias + residual -----------------------
__global__ void proj_kernel(const __bf16* __restrict__ A, const __bf16* __restrict__ WT,
                            const float* __restrict__ bias, const float* __restrict__ resid,
                            float* __restrict__ out) {
    int wid  = (blockIdx.x * blockDim.x + threadIdx.x) >> 5;
    int lane = threadIdx.x & 31;
    int half = lane >> 4, r = lane & 15;
    size_t m0 = (size_t)(wid >> 4) * 16;
    int n0 = (wid & 15) * 16;
    const __bf16* arow = A  + (m0 + r) * CDIM + 8 * half;
    const __bf16* brow = WT + (size_t)(n0 + r) * CDIM + 8 * half;
    f32x8 c = zero8();
#pragma unroll
    for (int kk = 0; kk < 8; ++kk)
        c = wmma_bf16(load_frag(arow + kk * 32), load_frag(brow + kk * 32), c);
    float bv = bias[n0 + r];
#pragma unroll
    for (int i = 0; i < 8; ++i) {
        size_t idx = (m0 + i + 8 * half) * CDIM + n0 + r;
        out[idx] = c[i] + bv + resid[idx];
    }
}

extern "C" void kernel_launch(void* const* d_in, const int* in_sizes, int n_in,
                              void* d_out, int out_size, void* d_ws, size_t ws_size,
                              hipStream_t stream) {
    const float* inputs  = (const float*)d_in[0];
    const float* context = (const float*)d_in[1];
    const float* Wq = (const float*)d_in[2];
    const float* bq = (const float*)d_in[3];
    const float* Wk = (const float*)d_in[4];
    const float* bk = (const float*)d_in[5];
    const float* Wv = (const float*)d_in[6];
    const float* bv = (const float*)d_in[7];
    const float* Wp = (const float*)d_in[8];
    const float* bp = (const float*)d_in[9];
    const float* gamma = (const float*)d_in[10];
    const float* beta  = (const float*)d_in[11];
    float* out = (float*)d_out;

    const size_t TOT = (size_t)NBATCH * N_SPATIAL * CDIM;  // 4,194,304 elements
    char* ws = (char*)d_ws;
    float*  xn_f32 = (float*)ws;  ws += TOT * 4;
    __bf16* xn_bf  = (__bf16*)ws; ws += TOT * 2;
    __bf16* ctx_bf = (__bf16*)ws; ws += TOT * 2;
    __bf16* qb     = (__bf16*)ws; ws += TOT * 2;
    __bf16* kb_    = (__bf16*)ws; ws += TOT * 2;
    __bf16* vtb    = (__bf16*)ws; ws += TOT * 2;
    __bf16* ob     = (__bf16*)ws; ws += TOT * 2;
    __bf16* WqT    = (__bf16*)ws; ws += 65536 * 2;
    __bf16* WkT    = (__bf16*)ws; ws += 65536 * 2;
    __bf16* WvT    = (__bf16*)ws; ws += 65536 * 2;
    __bf16* WpT    = (__bf16*)ws; ws += 65536 * 2;

    cvt_bf16_kernel<<<(int)(TOT / 256), 256, 0, stream>>>(context, ctx_bf, (int)TOT);
    transpose_w_kernel<<<1024, 256, 0, stream>>>(Wq, Wk, Wv, Wp, WqT, WkT, WvT, WpT);
    ln_kernel<<<NBATCH * N_SPATIAL, 256, 0, stream>>>(inputs, gamma, beta, xn_f32, xn_bf);
    qproj_kernel<<<2048, 256, 0, stream>>>(xn_bf, WqT, bq, qb);
    kvproj_kernel<<<2048, 256, 0, stream>>>(ctx_bf, WkT, WvT, bk, bv, kb_, vtb);
    flash_kernel<<<128, 256, 0, stream>>>(qb, kb_, vtb, ob);
    proj_kernel<<<2048, 256, 0, stream>>>(ob, WpT, bp, xn_f32, out);
}
